// HardVectorQuantizer_36945308680479
// MI455X (gfx1250) — compile-verified, matrix operand order checked by reading, not run on hardware
//
#include <hip/hip_runtime.h>
#include <math.h>

typedef __attribute__((ext_vector_type(16))) _Float16 v16h;
typedef __attribute__((ext_vector_type(8)))  _Float16 v8h;
typedef __attribute__((ext_vector_type(4)))  _Float16 v4h;
typedef __attribute__((ext_vector_type(8)))  float    v8f;
typedef __attribute__((ext_vector_type(4)))  float    v4f;
typedef __attribute__((ext_vector_type(4)))  int      v4i;

// Problem constants
#define NROWS   65536      // B*EMBED/CODE_DIM = 4096*1024/64
#define KCODES  4096
#define DDIM    64
#define TILES   (KCODES / 16)
#define CB_SCALE 4096.0f   // argmax invariant under uniform score scaling

// LDS staging: per buffer = 2048B codebook tile + 64B hn slice
#define TILE_HALVES (16 * DDIM)        // 1024 halves = 2048 B
#define BUF_HALVES  (TILE_HALVES + 32) // + 64 B hn    = 2112 B
#define NBUF        4
#define COPIES_PER_TILE 5              // 4 full-wave b128 groups + 1 masked

// d_out element offsets (float*), outputs concatenated in return order:
// quantized(4096*1024) | commit_loss(1) | indices(65536) | perplexity(1)
#define OUT_Q_OFF    0
#define OUT_LOSS_OFF 4194304
#define OUT_IDX_OFF  4194305
#define OUT_PERP_OFF 4259841

// d_ws byte offsets
#define WS_CB16_OFF  0                     // 4096*64 halves = 524288 B
#define WS_HN_OFF    524288                // 4096 f32      = 16384 B
#define WS_CNT_OFF   540672                // 4096 u32      = 16384 B
#define WS_LOSS_OFF  557056                // 1 f32

// ---- CDNA5 async global->LDS path (guarded; falls back to sync copy) ----
#if defined(__has_builtin)
#if __has_builtin(__builtin_amdgcn_global_load_async_to_lds_b128)
#define HAVE_ASYNC_LDS 1
#endif
#if __has_builtin(__builtin_amdgcn_s_wait_asynccnt)
#define HAVE_WAIT_ASYNC 1
#endif
#endif

__device__ __forceinline__ void async_cp16(const void* g, void* l) {
#if defined(HAVE_ASYNC_LDS)
  // prototype (from clang diagnostic): (v4i AS1*, v4i AS3*, i32 offset, i32 cpol)
  __builtin_amdgcn_global_load_async_to_lds_b128(
      (__attribute__((address_space(1))) v4i*)(g),
      (__attribute__((address_space(3))) v4i*)(l), 0, 0);
#else
  *(v4f*)l = *(const v4f*)g;   // synchronous fallback (regs + ds_store)
#endif
}

// Unconditional partial wait: non-producer waves have ASYNCcnt==0 and pass
// for free, so no exec-mask branch is needed around it. Threshold must be a
// compile-time constant (builtin requires an ICE), hence the template param.
template <int N>
__device__ __forceinline__ void wait_async_le() {
#if defined(HAVE_WAIT_ASYNC)
  __builtin_amdgcn_s_wait_asynccnt(N);
#elif defined(HAVE_ASYNC_LDS)
  asm volatile("s_wait_asynccnt %0" :: "i"(N) : "memory");
#endif
}

__device__ __forceinline__ unsigned umax(unsigned a, unsigned b) {
  return a > b ? a : b;
}

__device__ __forceinline__ v16h cat8(v8h lo, v8h hi) {
  v16h r;
#pragma unroll
  for (int i = 0; i < 8; ++i) { r[i] = lo[i]; r[8 + i] = hi[i]; }
  return r;
}

// Load one B-operand chunk (this lane's 32 K-slice) from a f32 z-row:
// halves [0..7] = dims [0..8), halves [8..15] = dims [16..24) rel. to base.
__device__ __forceinline__ v16h load_b_f32(const float* base) {
  v4f f0 = *(const v4f*)(base);
  v4f f1 = *(const v4f*)(base + 4);
  v4f f2 = *(const v4f*)(base + 16);
  v4f f3 = *(const v4f*)(base + 20);
  v16h b;
#pragma unroll
  for (int i = 0; i < 4; ++i) {
    b[i]      = (_Float16)f0[i];
    b[4 + i]  = (_Float16)f1[i];
    b[8 + i]  = (_Float16)f2[i];
    b[12 + i] = (_Float16)f3[i];
  }
  return b;
}

// Producer wave copies one tile into LDS: 2048B codebook tile (128 x 16B,
// 4 per lane) + 64B hn slice (lanes 0..3). 5 async instructions total.
__device__ __forceinline__ void issue_tile(const _Float16* __restrict__ cb16,
                                           const float* __restrict__ hn,
                                           _Float16* sdst, int t, int lane) {
  const char* gt = (const char*)(cb16 + (size_t)t * TILE_HALVES);
  char*       l  = (char*)sdst;
#pragma unroll
  for (int j = 0; j < 4; ++j) {
    const int c = lane * 16 + j * 512;
    async_cp16(gt + c, l + c);
  }
  if (lane < 4)
    async_cp16((const char*)(hn + t * 16) + lane * 16,
               l + 2048 + lane * 16);
}

// Zero the histogram + loss accumulator (harness poisons ws with 0xAA).
__global__ void vq_init(unsigned* __restrict__ counts, float* __restrict__ lossAcc) {
  int t = blockIdx.x * blockDim.x + threadIdx.x;
  if (t < KCODES) counts[t] = 0u;
  if (t == KCODES) *lossAcc = 0.0f;
}

// Convert codebook to f16 (scaled by 4096) and compute 0.5*scale*|c_k|^2.
__global__ void vq_prep(const float* __restrict__ cb,
                        _Float16* __restrict__ cb16,
                        float* __restrict__ hn) {
  int k = blockIdx.x * blockDim.x + threadIdx.x;   // one code per thread
  const float* src = cb + (size_t)k * DDIM;
  _Float16*    dst = cb16 + (size_t)k * DDIM;
  float s = 0.0f;
#pragma unroll
  for (int j = 0; j < 16; ++j) {
    v4f v = *(const v4f*)(src + 4 * j);
    v4h h;
#pragma unroll
    for (int i = 0; i < 4; ++i) {
      s += v[i] * v[i];
      h[i] = (_Float16)(v[i] * CB_SCALE);
    }
    *(v4h*)(dst + 4 * j) = h;
  }
  hn[k] = 0.5f * CB_SCALE * s;
}

// Main fused GEMM + argmax + gather + loss kernel.
// 512 blocks x 256 threads (8 waves). Each wave owns 16 z-rows; the block
// streams codebook tiles through a 4-deep LDS ring via async global->LDS
// copies (prefetch distance 2, partial asynccnt waits), and every wave runs
// two chained v_wmma_f32_16x16x32_f16 per tile. Argmax uses packed u32 keys:
// ordered-float[31:12] | (4095-code)[11:0], reduced with a parallel
// v_max_u32 tree (lowest-index tie-break for free).
__global__ void __launch_bounds__(256)
vq_main(const float* __restrict__ z, const float* __restrict__ cb,
        const _Float16* __restrict__ cb16, const float* __restrict__ hn,
        float* __restrict__ qOut, float* __restrict__ idxOut,
        unsigned* __restrict__ counts, float* __restrict__ lossAcc) {
  __shared__ _Float16 stile[NBUF][BUF_HALVES];   // 4 x 2112B ring

  const int lane    = threadIdx.x & 31;
  const int wave    = threadIdx.x >> 5;
  const int half    = lane >> 4;          // 0: codes 0-7 of tile, 1: codes 8-15
  const int n       = lane & 15;          // z-row within the wave's tile
  const int rowBase = blockIdx.x * 128 + wave * 16;
  const int row     = rowBase + n;
  const bool producer = (wave == 0);

  // ---- Load B operands (this lane's z-row, f32 -> f16), reused 256x ----
  const float* zr = z + (size_t)row * DDIM;
  const v16h blo = load_b_f32(zr + half * 8);        // dims 0..31 slice
  const v16h bhi = load_b_f32(zr + 32 + half * 8);   // dims 32..63 slice

  // ---- prologue: stage tiles 0 and 1 ----
  if (producer) {
    issue_tile(cb16, hn, stile[0], 0, lane);
    issue_tile(cb16, hn, stile[1], 1, lane);
  }
  wait_async_le<COPIES_PER_TILE>();   // tile 0 landed (tile 1 may be in flight)
  __syncthreads();

  unsigned best = 0u;   // any real ordered key beats 0

  for (int t = 0; t < TILES; ++t) {
    const int p = t & (NBUF - 1);
    if (producer && (t + 2 < TILES))
      issue_tile(cb16, hn, stile[(t + 2) & (NBUF - 1)], t + 2, lane);

    // ---- A operand from LDS (row-major f16, ds_load_b128 chunks) ----
    const _Float16* ab = &stile[p][n * DDIM + half * 8];
    v8h a0 = *(const v8h*)(ab);        // K {0..7} / {8..15}
    v8h a1 = *(const v8h*)(ab + 16);   // K {16..23} / {24..31}
    v8h a2 = *(const v8h*)(ab + 32);   // K {32..39} / {40..47}
    v8h a3 = *(const v8h*)(ab + 48);   // K {48..55} / {56..63}
    v16h alo = cat8(a0, a1);
    v16h ahi = cat8(a2, a3);

    v8f c = {};
    c = __builtin_amdgcn_wmma_f32_16x16x32_f16(false, alo, false, blo,
                                               (short)0, c, false, false);
    c = __builtin_amdgcn_wmma_f32_16x16x32_f16(false, ahi, false, bhi,
                                               (short)0, c, false, false);

    // ---- packed-key argmax (parallel, no vcc serialization) ----
    const float* hb = (const float*)&stile[p][TILE_HALVES];   // 16 hn floats
    v4f h0 = *(const v4f*)(hb + half * 8);
    v4f h1 = *(const v4f*)(hb + half * 8 + 4);
    const int kb = t * 16 + half * 8;
    const unsigned inv0 = 4095u - (unsigned)kb;
    unsigned k[8];
#pragma unroll
    for (int i = 0; i < 8; ++i) {
      float sc = c[i] - ((i < 4) ? h0[i] : h1[i - 4]);
      unsigned u = __float_as_uint(sc);
      u ^= ((unsigned)((int)u >> 31)) | 0x80000000u;      // total order
      k[i] = (u & 0xFFFFF000u) | (inv0 - (unsigned)i);    // score|inv(code)
    }
    unsigned m0 = umax(k[0], k[1]), m1 = umax(k[2], k[3]);
    unsigned m2 = umax(k[4], k[5]), m3 = umax(k[6], k[7]);
    best = umax(best, umax(umax(m0, m1), umax(m2, m3)));

    // tile t+1 complete (async loads retire in order); t+2 may stay in flight
    wait_async_le<COPIES_PER_TILE>();
    __syncthreads();              // publish buffer / protect ring reuse
  }

  // ---- combine the two lane-halves (codes 0-7 vs 8-15 of each tile) ----
  best = umax(best, (unsigned)__shfl_xor((int)best, 16, 32));
  const int bestIdx = 4095 - (int)(best & 0xFFFu);

  if (lane < 16) {
    idxOut[row] = (float)bestIdx;          // encoding_indices (stored as f32)
    atomicAdd(&counts[bestIdx], 1u);
  }

  // ---- gather quantized rows (coalesced: 32 lanes x float2 = one row) ----
  float lsum = 0.0f;
#pragma unroll 4
  for (int r = 0; r < 16; ++r) {
    int idxr = __shfl(bestIdx, r, 32);     // lanes 0..15 hold final results
    const int rrow = rowBase + r;
    float2 q  = *(const float2*)(cb + (size_t)idxr * DDIM + lane * 2);
    float2 zz = *(const float2*)(z + (size_t)rrow * DDIM + lane * 2);
    float dx = zz.x - q.x, dy = zz.y - q.y;
    lsum += dx * dx + dy * dy;
    *(float2*)(qOut + (size_t)rrow * DDIM + lane * 2) = q;
  }
  // wave reduce the commit-loss partial
#pragma unroll
  for (int m = 16; m >= 1; m >>= 1) lsum += __shfl_xor(lsum, m, 32);
  if (lane == 0) atomicAdd(lossAcc, lsum);
}

// Finalize commit_loss and perplexity.
__global__ void vq_final(const unsigned* __restrict__ counts,
                         const float* __restrict__ lossAcc,
                         float* __restrict__ out) {
  __shared__ float sh[256];
  float h = 0.0f;
  for (int b = threadIdx.x; b < KCODES; b += 256) {
    float p = (float)counts[b] * (1.0f / (float)NROWS);
    h += p * logf(p + 1e-10f);
  }
  sh[threadIdx.x] = h;
  __syncthreads();
  for (int s = 128; s > 0; s >>= 1) {
    if (threadIdx.x < s) sh[threadIdx.x] += sh[threadIdx.x + s];
    __syncthreads();
  }
  if (threadIdx.x == 0) {
    out[OUT_LOSS_OFF] = (*lossAcc) * (0.1f / (float)(NROWS * DDIM));
    out[OUT_PERP_OFF] = expf(-sh[0]);
  }
}

extern "C" void kernel_launch(void* const* d_in, const int* in_sizes, int n_in,
                              void* d_out, int out_size, void* d_ws, size_t ws_size,
                              hipStream_t stream) {
  const float* z  = (const float*)d_in[0];   // (4096, 1024) f32
  const float* cb = (const float*)d_in[1];   // (4096, 64)   f32
  char* ws = (char*)d_ws;
  _Float16* cb16   = (_Float16*)(ws + WS_CB16_OFF);
  float*    hn     = (float*)(ws + WS_HN_OFF);
  unsigned* counts = (unsigned*)(ws + WS_CNT_OFF);
  float*    lossA  = (float*)(ws + WS_LOSS_OFF);
  float*    out    = (float*)d_out;

  vq_init<<<(KCODES + 256) / 256, 256, 0, stream>>>(counts, lossA);
  vq_prep<<<KCODES / 256, 256, 0, stream>>>(cb, cb16, hn);
  vq_main<<<NROWS / 128, 256, 0, stream>>>(z, cb, cb16, hn,
                                           out + OUT_Q_OFF, out + OUT_IDX_OFF,
                                           counts, lossA);
  vq_final<<<1, 256, 0, stream>>>(counts, lossA, out);
}
